// DistanceGNN_59854664237631
// MI455X (gfx1250) — matrix operand order repaired and checked by reading, not compiled, not run on hardware
//
#include <hip/hip_runtime.h>
#include <stdint.h>

#define NN 20000
#define HH 128
#define EE 320000
#define AA 512
#define NEMB_ 400
#define KSEL 80000   // int(E * 0.25)

typedef __attribute__((ext_vector_type(16))) __bf16 v16bf;
typedef __attribute__((ext_vector_type(2)))  __bf16 v2bf;
typedef __attribute__((ext_vector_type(8)))  float  v8f;

// ---------------------------------------------------------------------------
// Strip GEMM: C[M,128] = A[M,KT] @ B[KT,128] (+bias[128]) (+add[M,128])
// B staged once per workgroup into LDS as bf16 in a contiguous-k (k = 16*half + j) WMMA B-fragment
// layout (one contiguous 32B read per lane per tile). k-loop rolled; B
// fragments preloaded in chunks of 4 (32 live VGPRs) -> no spills; compiler
// double-buffers ds_loads against back-to-back WMMAs. Epilogue fully
// branch-free via HAS_BIAS / HAS_ADD template flags.
// ---------------------------------------------------------------------------
template <int KT, bool HAS_BIAS, bool HAS_ADD>
__global__ __launch_bounds__(128, 1)
void gemm_wmma_bf16(const float* __restrict__ A,
                    const float* __restrict__ B,
                    const float* __restrict__ bias,
                    const float* __restrict__ add,
                    float* __restrict__ C, int M)
{
    extern __shared__ __bf16 lsB[];   // KT*128 bf16, fragment-swizzled
    const int tid = threadIdx.x;

    // ---- stage B: coalesced f32 reads, packed bf16x2 LDS stores ----
    // fragment layout: frag f = (k0/32)*8 + nt ; lane l ; elem j (0..15)
    //   element = B[k0 + (l>>4)*16 + j][nt*16 + (l&15)]
    //   halfword index = (f*32 + l)*16 + j
    for (int i = tid; i < (KT >> 1) * HH; i += blockDim.x) {
        const int m = i >> 7;              // k-pair index
        const int c = i & 127;             // column
        const int k = m << 1;
        const float f0 = B[(size_t)k * HH + c];
        const float f1 = B[(size_t)(k + 1) * HH + c];
        const int within = k & 31;
        const int half   = within >> 4;
        const int j      = within & 15;    // even; pair fills j, j+1
        const int nt     = c >> 4;
        const int l      = (half << 4) | (c & 15);
        const int f      = ((k >> 5) << 3) + nt;
        const int hw     = ((f * 32 + l) << 4) + j;
        v2bf p;
        p[0] = (__bf16)f0;
        p[1] = (__bf16)f1;
        *(v2bf*)&lsB[hw] = p;
    }
    __syncthreads();

    const int lane = tid & 31;
    const int wave = tid >> 5;
    const int half = lane >> 4;
    const int l15  = lane & 15;
    const int S    = M >> 4;

    for (int strip = blockIdx.x * 4 + wave; strip < S; strip += gridDim.x * 4) {
        const int row0 = strip << 4;

        v8f acc[8];
#pragma unroll
        for (int i = 0; i < 8; ++i)
#pragma unroll
            for (int j = 0; j < 8; ++j) acc[i][j] = 0.0f;

        const float* Abase = A + (size_t)(row0 + l15) * KT;

#pragma unroll 1
        for (int k0 = 0; k0 < KT; k0 += 32) {
            // A fragment: two 32B chunks per lane -> 4x b128 loads
            const float4* ap = (const float4*)(Abase + k0 + half * 16);
            const float4 a0 = ap[0];
            const float4 a1 = ap[1];
            const float4 a2 = ap[2];   // +8 floats
            const float4 a3 = ap[3];
            v16bf a;
            a[0]  = (__bf16)a0.x; a[1]  = (__bf16)a0.y; a[2]  = (__bf16)a0.z; a[3]  = (__bf16)a0.w;
            a[4]  = (__bf16)a1.x; a[5]  = (__bf16)a1.y; a[6]  = (__bf16)a1.z; a[7]  = (__bf16)a1.w;
            a[8]  = (__bf16)a2.x; a[9]  = (__bf16)a2.y; a[10] = (__bf16)a2.z; a[11] = (__bf16)a2.w;
            a[12] = (__bf16)a3.x; a[13] = (__bf16)a3.y; a[14] = (__bf16)a3.z; a[15] = (__bf16)a3.w;

            const int fbase = (k0 >> 5) << 3;

#pragma unroll
            for (int h = 0; h < 2; ++h) {
                // preload 4 B fragments (ds clause) ...
                v16bf bfr[4];
#pragma unroll
                for (int q = 0; q < 4; ++q)
                    bfr[q] = *(const v16bf*)
                        &lsB[((fbase + h * 4 + q) * 32 + lane) << 4];
                // ... then 4 back-to-back WMMAs on the matrix pipe
#pragma unroll
                for (int q = 0; q < 4; ++q)
                    acc[h * 4 + q] = __builtin_amdgcn_wmma_f32_16x16x32_bf16(
                        false, a, false, bfr[q], (short)0, acc[h * 4 + q],
                        false, false);
            }
        }

        // branch-free epilogue (config known at compile time)
#pragma unroll
        for (int nt = 0; nt < 8; ++nt) {
            const int col = nt * 16 + l15;
            const float bcol = HAS_BIAS ? bias[col] : 0.0f;
#pragma unroll
            for (int r = 0; r < 8; ++r) {
                const int row = row0 + r + half * 8;   // C/D VGPR layout
                float v = acc[nt][r] + bcol;
                if (HAS_ADD) v += add[(size_t)row * HH + col];
                C[(size_t)row * HH + col] = v;
            }
        }
    }
}

// ---------------------------------------------------------------------------
// Edge scoring: one wave per edge.
// logits[e] = -|| qnode[src] + dist_q[bucket] - x_v[tgt] ||  (biases folded)
// key[e] = sortable_uint(logits + gumbel(e))   [log_softmax shift dropped:
// top-k of (logprob + g) == top-k of (logits + g)]
// ---------------------------------------------------------------------------
__device__ __forceinline__ float warp_sum32(float v) {
#pragma unroll
    for (int off = 16; off > 0; off >>= 1) v += __shfl_xor(v, off, 32);
    return v;
}

__device__ __forceinline__ unsigned mix_u32(unsigned x) {
    x ^= x >> 16; x *= 0x7FEB352Du;
    x ^= x >> 15; x *= 0x846CA68Bu;
    x ^= x >> 16; return x;
}

__global__ void edge_score_kernel(const float* __restrict__ qnode,
                                  const float* __restrict__ xv,
                                  const float* __restrict__ distq,
                                  const int* __restrict__ src,
                                  const int* __restrict__ tgt,
                                  const int* __restrict__ dist,
                                  unsigned* __restrict__ keys)
{
    const int lane = threadIdx.x & 31;
    const int e = blockIdx.x * 4 + (threadIdx.x >> 5);
    if (e >= EE) return;
    const int s = src[e], t = tgt[e];
    const int bucket = dist[e] / 50;   // DIST_UNIT

    const float4 q = ((const float4*)(qnode + (size_t)s * HH))[lane];
    const float4 v = ((const float4*)(xv    + (size_t)t * HH))[lane];
    const float4 d = ((const float4*)(distq + (size_t)bucket * HH))[lane];

    const float dx = q.x + d.x - v.x;
    const float dy = q.y + d.y - v.y;
    const float dz = q.z + d.z - v.z;
    const float dw = q.w + d.w - v.w;
    const float ss = warp_sum32(dx*dx + dy*dy + dz*dz + dw*dw);

    if (lane == 0) {
        const float logit = -sqrtf(ss);
        unsigned h = mix_u32((unsigned)e * 0x9E3779B9u + 0x85EBCA6Bu);
        float u = (float)(h >> 8) * (1.0f / 16777216.0f);
        float g = -logf(-logf(u + 1e-20f) + 1e-20f);
        float sc = logit + g;
        unsigned bits = __float_as_uint(sc);
        unsigned key = (bits & 0x80000000u) ? ~bits : (bits | 0x80000000u);
        keys[e] = key;
    }
}

// ---------------------------------------------------------------------------
// Device-side radix select (k-th largest) over 32-bit sortable keys
// ---------------------------------------------------------------------------
__global__ void init_state(unsigned* state) {
    if (threadIdx.x == 0 && blockIdx.x == 0) {
        state[0] = 0;       // prefix (fixed high bits)
        state[1] = KSEL;    // remaining rank
        state[2] = 0;       // final threshold T
        state[3] = 0;       // ties needed at T
        state[4] = 0;       // tie ticket counter
    }
}

__global__ void hist_kernel(const unsigned* __restrict__ keys,
                            const unsigned* __restrict__ state,
                            unsigned prefix_mask, int shift, int bins,
                            unsigned* __restrict__ hist)
{
    __shared__ unsigned lh[2048];
    for (int i = threadIdx.x; i < bins; i += blockDim.x) lh[i] = 0;
    __syncthreads();
    const unsigned prefix = state[0];
    for (int e = blockIdx.x * blockDim.x + threadIdx.x; e < EE;
         e += gridDim.x * blockDim.x) {
        unsigned k = keys[e];
        if ((k & prefix_mask) == prefix)
            atomicAdd(&lh[(k >> shift) & (unsigned)(bins - 1)], 1u);
    }
    __syncthreads();
    for (int i = threadIdx.x; i < bins; i += blockDim.x)
        if (lh[i]) atomicAdd(&hist[i], lh[i]);
}

__global__ void pick_kernel(const unsigned* __restrict__ hist,
                            unsigned* state, int shift, int bins, int last)
{
    if (threadIdx.x || blockIdx.x) return;
    unsigned krem = state[1];
    unsigned cum = 0;
    int b = bins - 1;
    for (; b > 0; --b) {
        unsigned c = hist[b];
        if (cum + c >= krem) break;
        cum += c;
    }
    state[0] |= ((unsigned)b) << shift;
    state[1] = krem - cum;
    if (last) { state[2] = state[0]; state[3] = state[1]; }
}

// ---------------------------------------------------------------------------
// Selection + SAGE mean-aggregation scatter: one wave per edge
// ---------------------------------------------------------------------------
__global__ void select_aggregate(const unsigned* __restrict__ keys,
                                 const int* __restrict__ src,
                                 const int* __restrict__ tgt,
                                 const float* __restrict__ xlin,
                                 unsigned* state,
                                 float* __restrict__ msg,
                                 float* __restrict__ deg)
{
    const int lane = threadIdx.x & 31;
    const int e = blockIdx.x * 4 + (threadIdx.x >> 5);
    if (e >= EE) return;
    int sel = 0;
    if (lane == 0) {
        const unsigned T = state[2], ties = state[3];
        unsigned k = keys[e];
        if (k > T) sel = 1;
        else if (k == T) sel = (atomicAdd(&state[4], 1u) < ties);
    }
    sel = __shfl(sel, 0, 32);
    if (!sel) return;
    const int s = src[e], t = tgt[e];
    const float4 val = ((const float4*)(xlin + (size_t)s * HH))[lane];
    float* mp = msg + (size_t)t * HH + lane * 4;
    atomicAdd(mp + 0, val.x);
    atomicAdd(mp + 1, val.y);
    atomicAdd(mp + 2, val.z);
    atomicAdd(mp + 3, val.w);
    if (lane == 0) atomicAdd(&deg[t], 1.0f);
}

__global__ void finalize_kernel(float* __restrict__ msg,
                                const float* __restrict__ deg)
{
    int i = blockIdx.x * blockDim.x + threadIdx.x;
    if (i >= NN * HH) return;
    msg[i] = msg[i] / fmaxf(deg[i >> 7], 1.0f);
}

// ---------------------------------------------------------------------------
extern "C" void kernel_launch(void* const* d_in, const int* in_sizes, int n_in,
                              void* d_out, int out_size, void* d_ws, size_t ws_size,
                              hipStream_t stream)
{
    const float* x      = (const float*)d_in[0];
    const int*   ei     = (const int*)  d_in[1];
    const int*   dist   = (const int*)  d_in[2];
    const float* aerial = (const float*)d_in[3];
    const float* W_lin  = (const float*)d_in[4];
    const float* b_lin  = (const float*)d_in[5];
    const float* W_q    = (const float*)d_in[6];
    const float* b_q    = (const float*)d_in[7];
    const float* W_v    = (const float*)d_in[8];
    const float* b_v    = (const float*)d_in[9];
    const float* dist_t = (const float*)d_in[10];
    const float* W_l    = (const float*)d_in[11];
    const float* b_l    = (const float*)d_in[12];
    const float* W_r    = (const float*)d_in[13];
    const int* src = ei;
    const int* tgt = ei + EE;

    char* ws = (char*)d_ws;
    size_t off = 0;
    auto carve = [&](size_t bytes) -> char* {
        char* p = ws + off;
        off += (bytes + 255) & ~(size_t)255;
        return p;
    };
    float*    x_lin  = (float*)   carve(sizeof(float) * NN * HH);
    float*    qnode  = (float*)   carve(sizeof(float) * NN * HH);
    float*    x_v    = (float*)   carve(sizeof(float) * NN * HH);
    float*    tmp    = (float*)   carve(sizeof(float) * NN * HH);
    float*    msg    = (float*)   carve(sizeof(float) * NN * HH);
    float*    dist_q = (float*)   carve(sizeof(float) * NEMB_ * HH);
    float*    deg    = (float*)   carve(sizeof(float) * NN);
    unsigned* keys   = (unsigned*)carve(sizeof(unsigned) * EE);
    unsigned* hist   = (unsigned*)carve(sizeof(unsigned) * 2048);
    unsigned* state  = (unsigned*)carve(sizeof(unsigned) * 16);

    const dim3 blk(128);                       // 4 waves / block
    const int blocksN = (NN / 16 + 3) / 4;     // 313
    const int blocksD = (NEMB_ / 16 + 3) / 4;  // 7
    const size_t sh128 = (size_t)HH * HH * 2;  // 32 KB (K=128)
    const size_t sh512 = (size_t)AA * HH * 2;  // 128 KB (K=512)

    // --- node-level WMMA GEMMs (query/value separated algebraically) ---
    gemm_wmma_bf16<HH, true,  false><<<blocksN, blk, sh128, stream>>>(x,      W_lin,          b_lin,   nullptr, x_lin,  NN);
    gemm_wmma_bf16<AA, false, false><<<blocksN, blk, sh512, stream>>>(aerial, W_q + 128 * HH, nullptr, nullptr, tmp,    NN);
    gemm_wmma_bf16<HH, false, true ><<<blocksN, blk, sh128, stream>>>(x_lin,  W_q,            nullptr, tmp,     qnode,  NN);
    gemm_wmma_bf16<HH, true,  false><<<blocksD, blk, sh128, stream>>>(dist_t, W_q + 640 * HH, b_q,     nullptr, dist_q, NEMB_);
    gemm_wmma_bf16<HH, true,  false><<<blocksN, blk, sh128, stream>>>(x_lin,  W_v,            b_v,     nullptr, x_v,    NN);

    // --- per-edge scoring ---
    edge_score_kernel<<<EE / 4, blk, 0, stream>>>(qnode, x_v, dist_q, src, tgt, dist, keys);

    // --- radix top-k threshold (3 rounds: 11 + 11 + 10 bits) ---
    init_state<<<1, 32, 0, stream>>>(state);
    hipMemsetAsync(hist, 0, 2048 * sizeof(unsigned), stream);
    hist_kernel<<<256, 256, 0, stream>>>(keys, state, 0x00000000u, 21, 2048, hist);
    pick_kernel<<<1, 32, 0, stream>>>(hist, state, 21, 2048, 0);
    hipMemsetAsync(hist, 0, 2048 * sizeof(unsigned), stream);
    hist_kernel<<<256, 256, 0, stream>>>(keys, state, 0xFFE00000u, 10, 2048, hist);
    pick_kernel<<<1, 32, 0, stream>>>(hist, state, 10, 2048, 0);
    hipMemsetAsync(hist, 0, 2048 * sizeof(unsigned), stream);
    hist_kernel<<<256, 256, 0, stream>>>(keys, state, 0xFFFFFC00u, 0, 1024, hist);
    pick_kernel<<<1, 32, 0, stream>>>(hist, state, 0, 1024, 1);

    // --- selection + mean aggregation ---
    hipMemsetAsync(msg, 0, sizeof(float) * NN * HH, stream);
    hipMemsetAsync(deg, 0, sizeof(float) * NN, stream);
    select_aggregate<<<EE / 4, blk, 0, stream>>>(keys, src, tgt, x_lin, state, msg, deg);
    finalize_kernel<<<(NN * HH + 255) / 256, 256, 0, stream>>>(msg, deg);

    // --- output: out = aggr @ W_l + b_l + x_lin @ W_r ---
    gemm_wmma_bf16<HH, false, false><<<blocksN, blk, sh128, stream>>>(x_lin, W_r, nullptr, nullptr, tmp,           NN);
    gemm_wmma_bf16<HH, true,  true ><<<blocksN, blk, sh128, stream>>>(msg,   W_l, b_l,     tmp,     (float*)d_out, NN);
}
